// GAT_10969346474857
// MI455X (gfx1250) — matrix operand order repaired and checked by reading, not compiled, not run on hardware
//
#include <hip/hip_runtime.h>

// ---------------------------------------------------------------------------
// GAT forward for MI455X (gfx1250, wave32).
//   layer GEMMs      : v_wmma_f32_16x16x32_bf16 (f32 accumulate), direct
//                      global fragment loads (operands are L2-resident)
//   edge phase       : edge-parallel, (edge,head) per thread, L2-served
//                      gathers + global_prefetch, float atomics for
//                      segment max / sum / scatter
//   classifier       : per-node dot products + log-softmax, coalesced LDS
// ---------------------------------------------------------------------------

typedef __attribute__((ext_vector_type(16))) __bf16     v16bf;
typedef __attribute__((ext_vector_type(8)))  float      v8f;
typedef __attribute__((ext_vector_type(4)))  unsigned   v4u;

union FragU { v4u u[2]; v16bf v; };

#define GAT_N     50000
#define GAT_E     800000
#define GAT_FIN   128
#define GAT_DHID  96
#define GAT_H     8
#define GAT_DH    12
#define GAT_NCLS  40
#define NEG_SLOPE 0.2f

// ------------------------------ conversions --------------------------------

__global__ void k_cvt_bf16(const float* __restrict__ src, __bf16* __restrict__ dst, int n) {
    int i = blockIdx.x * blockDim.x + threadIdx.x;
    if (i < n) dst[i] = (__bf16)src[i];
}

// W is [K,96] row-major fp32 -> Wt is [96][K] row-major bf16 (i.e. transposed)
__global__ void k_cvt_w_t(const float* __restrict__ W, __bf16* __restrict__ Wt, int K) {
    int i = blockIdx.x * blockDim.x + threadIdx.x;
    int total = 96 * K;
    if (i >= total) return;
    int n = i / K;
    int k = i - n * K;
    Wt[i] = (__bf16)W[k * 96 + n];
}

// ------------------------------ WMMA GEMM ----------------------------------
// C[M,96] = A[M,K](bf16) @ B[K,96]; B given transposed as Bt[96][K] bf16.
// 256 threads = 8 waves; wave w -> rows [blk*128 + w*16, +16), 6 col tiles.
__global__ __launch_bounds__(256) void k_gemm_att(const __bf16* __restrict__ A,
                                                  const __bf16* __restrict__ Bt,
                                                  float* __restrict__ C,
                                                  int M, int K) {
    const int lane = threadIdx.x & 31;
    const int wave = threadIdx.x >> 5;
    const int row0 = blockIdx.x * 128 + wave * 16;
    const int mloc = lane & 15;
    const int hi   = lane >> 4;      // 0: lanes 0-15, 1: lanes 16-31
    int arow = row0 + mloc;
    if (arow >= M) arow = M - 1;     // clamp (stores are guarded; keeps EXEC full)
    const int kbA = hi * 8;          // A 16-bit layout: K kb..kb+7 and 16+kb..+7
    const int kbB = hi * 8;          // B 16-bit layout: K kb..kb+7 and 16+kb..+7

    v8f acc[6];
#pragma unroll
    for (int t = 0; t < 6; ++t) acc[t] = (v8f){0.f,0.f,0.f,0.f,0.f,0.f,0.f,0.f};

    for (int k0 = 0; k0 < K; k0 += 32) {
        FragU fa;
        const __bf16* pa = A + (size_t)arow * K + k0 + kbA;
        fa.u[0] = *(const v4u*)(pa);        // K = kb .. kb+7
        fa.u[1] = *(const v4u*)(pa + 16);   // K = 16+kb .. 16+kb+7
#pragma unroll
        for (int t = 0; t < 6; ++t) {
            FragU fb;
            const __bf16* pb = Bt + (size_t)(t * 16 + mloc) * K + k0 + kbB;
            fb.u[0] = *(const v4u*)(pb);      // K = kb .. kb+7
            fb.u[1] = *(const v4u*)(pb + 16); // K = 16+kb .. 16+kb+7
            acc[t] = __builtin_amdgcn_wmma_f32_16x16x32_bf16(
                false, fa.v, false, fb.v, (short)0, acc[t], false, false);
        }
    }

#pragma unroll
    for (int t = 0; t < 6; ++t) {
#pragma unroll
        for (int r = 0; r < 8; ++r) {
            int row = row0 + r + hi * 8;     // C layout: lanes 0-15 -> M=r, 16-31 -> M=r+8
            if (row < M) C[(size_t)row * 96 + t * 16 + mloc] = acc[t][r];
        }
    }
}

// -------------------------- attention coefficients -------------------------

__global__ void k_alpha(const float* __restrict__ hlin,
                        const float* __restrict__ a_src,
                        const float* __restrict__ a_dst,
                        float* __restrict__ asn, float* __restrict__ adn, int Nn) {
    int i = blockIdx.x * blockDim.x + threadIdx.x;
    if (i >= Nn * GAT_H) return;
    int n = i >> 3, h = i & 7;
    const float* hp = hlin + (size_t)n * GAT_DHID + h * GAT_DH;
    const float* ap = a_src + h * GAT_DH;
    const float* dp = a_dst + h * GAT_DH;
    float s = 0.f, d = 0.f;
#pragma unroll
    for (int j = 0; j < GAT_DH; ++j) { float v = hp[j]; s += v * ap[j]; d += v * dp[j]; }
    asn[i] = s; adn[i] = d;
}

// ------------------------------- edge phase --------------------------------

__global__ void k_init(float* __restrict__ agg, float* __restrict__ m,
                       float* __restrict__ z, int Nn) {
    int i = blockIdx.x * blockDim.x + threadIdx.x;
    if (i < Nn * GAT_DHID) agg[i] = 0.f;
    if (i < Nn * GAT_H) { m[i] = -INFINITY; z[i] = 0.f; }
}

__device__ __forceinline__ float edge_logit(const float* asn, const float* adn,
                                            int s, int d, int h) {
    float v = asn[s * GAT_H + h] + adn[d * GAT_H + h];
    return v > 0.f ? v : NEG_SLOPE * v;
}

// float atomic max via sign-split (correct with -inf init)
__device__ __forceinline__ void atomic_max_f32(float* a, float v) {
    if (v >= 0.f) atomicMax((int*)a, __float_as_int(v));
    else          atomicMin((unsigned int*)a, __float_as_uint(v));
}

__global__ void k_edge_max(const int* __restrict__ src, const int* __restrict__ dst,
                           const float* __restrict__ asn, const float* __restrict__ adn,
                           float* __restrict__ m, int Ee) {
    int i = blockIdx.x * blockDim.x + threadIdx.x;
    if (i >= Ee * GAT_H) return;
    int e = i >> 3, h = i & 7;
    int s = src[e], d = dst[e];
    atomic_max_f32(&m[d * GAT_H + h], edge_logit(asn, adn, s, d, h));
}

__global__ void k_edge_sum(const int* __restrict__ src, const int* __restrict__ dst,
                           const float* __restrict__ asn, const float* __restrict__ adn,
                           const float* __restrict__ m, float* __restrict__ z, int Ee) {
    int i = blockIdx.x * blockDim.x + threadIdx.x;
    if (i >= Ee * GAT_H) return;
    int e = i >> 3, h = i & 7;
    int s = src[e], d = dst[e];
    float p = __expf(edge_logit(asn, adn, s, d, h) - m[d * GAT_H + h]);
    atomicAdd(&z[d * GAT_H + h], p);
}

__global__ void k_edge_agg(const int* __restrict__ src, const int* __restrict__ dst,
                           const float* __restrict__ asn, const float* __restrict__ adn,
                           const float* __restrict__ m, const float* __restrict__ z,
                           const float* __restrict__ hlin, float* __restrict__ agg, int Ee) {
    int i = blockIdx.x * blockDim.x + threadIdx.x;
    if (i >= Ee * GAT_H) return;
    int e = i >> 3, h = i & 7;
    int s = src[e], d = dst[e];
    // pull the message row toward the caches while we compute alpha
    const float* hrow = hlin + (size_t)s * GAT_DHID + h * GAT_DH;
    __builtin_prefetch(hrow, 0, 3);            // -> global_prefetch_b8
    float p = __expf(edge_logit(asn, adn, s, d, h) - m[d * GAT_H + h]);
    float alpha = p / (z[d * GAT_H + h] + 1e-16f);
    const float4* hs = (const float4*)hrow;
    float* ag = agg + (size_t)d * GAT_DHID + h * GAT_DH;
#pragma unroll
    for (int q = 0; q < 3; ++q) {
        float4 v = hs[q];
        atomicAdd(ag + q * 4 + 0, v.x * alpha);
        atomicAdd(ag + q * 4 + 1, v.y * alpha);
        atomicAdd(ag + q * 4 + 2, v.z * alpha);
        atomicAdd(ag + q * 4 + 3, v.w * alpha);
    }
}

// bias (+optional ReLU), write fp32 and/or bf16 copy for the next GEMM
__global__ void k_bias_act(const float* __restrict__ agg, const float* __restrict__ b,
                           float* __restrict__ out_f32, __bf16* __restrict__ out_bf,
                           int relu, int Nn) {
    int i = blockIdx.x * blockDim.x + threadIdx.x;
    if (i >= Nn * GAT_DHID) return;
    int c = i % GAT_DHID;
    float v = agg[i] + b[c];
    if (relu) v = fmaxf(v, 0.f);
    if (out_f32) out_f32[i] = v;
    if (out_bf)  out_bf[i]  = (__bf16)v;
}

// ----------------------- classifier + log-softmax --------------------------

#define OBLK 64
__global__ __launch_bounds__(OBLK) void k_out(const float* __restrict__ h,
                                              const float* __restrict__ Wout,
                                              const float* __restrict__ bout,
                                              float* __restrict__ out, int Nn) {
    __shared__ float sW[GAT_DHID * GAT_NCLS];   // 15360 B
    __shared__ float sb[GAT_NCLS];
    __shared__ float sh[OBLK * GAT_DHID];       // 24576 B
    const int tid = threadIdx.x;
    for (int i = tid; i < GAT_DHID * GAT_NCLS; i += OBLK) sW[i] = Wout[i];
    if (tid < GAT_NCLS) sb[tid] = bout[tid];
    // cooperative, fully coalesced stage of this block's 64 contiguous rows
    {
        const size_t gbase = (size_t)blockIdx.x * OBLK * GAT_DHID;
        const size_t gmax  = (size_t)Nn * GAT_DHID;
        for (int i = tid; i < OBLK * GAT_DHID; i += OBLK) {
            size_t g = gbase + i;
            sh[i] = (g < gmax) ? h[g] : 0.f;
        }
    }
    __syncthreads();
    const int node = blockIdx.x * OBLK + tid;
    if (node >= Nn) return;
    float logits[GAT_NCLS];
#pragma unroll
    for (int c = 0; c < GAT_NCLS; ++c) {
        float a = sb[c];
        for (int k = 0; k < GAT_DHID; ++k) a += sh[tid * GAT_DHID + k] * sW[k * GAT_NCLS + c];
        logits[c] = a;
    }
    float mx = logits[0];
#pragma unroll
    for (int c = 1; c < GAT_NCLS; ++c) mx = fmaxf(mx, logits[c]);
    float s = 0.f;
#pragma unroll
    for (int c = 0; c < GAT_NCLS; ++c) s += __expf(logits[c] - mx);
    float lse = mx + __logf(s);
    float* op = out + (size_t)node * GAT_NCLS;
#pragma unroll
    for (int c = 0; c < GAT_NCLS; ++c) op[c] = logits[c] - lse;
}

// ------------------------------- host side ---------------------------------

extern "C" void kernel_launch(void* const* d_in, const int* in_sizes, int n_in,
                              void* d_out, int out_size, void* d_ws, size_t ws_size,
                              hipStream_t stream) {
    (void)in_sizes; (void)n_in; (void)out_size; (void)ws_size;
    const int N = GAT_N, E = GAT_E;

    const float* x    = (const float*)d_in[0];
    const int*   esrc = (const int*)  d_in[1];
    const int*   edst = (const int*)  d_in[2];
    const float* W0   = (const float*)d_in[3];
    const float* as0  = (const float*)d_in[4];
    const float* ad0  = (const float*)d_in[5];
    const float* b0   = (const float*)d_in[6];
    const float* W1   = (const float*)d_in[7];
    const float* as1  = (const float*)d_in[8];
    const float* ad1  = (const float*)d_in[9];
    const float* b1   = (const float*)d_in[10];
    const float* Wout = (const float*)d_in[11];
    const float* bout = (const float*)d_in[12];
    float* out = (float*)d_out;

    char* ws = (char*)d_ws;
    size_t off = 0;
    auto take = [&](size_t bytes) -> void* {
        void* p = ws + off;
        off += (bytes + 255) & ~(size_t)255;
        return p;
    };
    __bf16* xb   = (__bf16*)take((size_t)N * GAT_FIN  * 2);
    __bf16* w0t  = (__bf16*)take((size_t)96 * GAT_FIN * 2);
    __bf16* w1t  = (__bf16*)take((size_t)96 * 96      * 2);
    float*  hlin = (float*) take((size_t)N * GAT_DHID * 4);
    float*  asn  = (float*) take((size_t)N * GAT_H    * 4);
    float*  adn  = (float*) take((size_t)N * GAT_H    * 4);
    float*  mbuf = (float*) take((size_t)N * GAT_H    * 4);
    float*  zbuf = (float*) take((size_t)N * GAT_H    * 4);
    float*  agg  = (float*) take((size_t)N * GAT_DHID * 4);
    __bf16* h0b  = (__bf16*)take((size_t)N * GAT_DHID * 2);
    float*  h1   = (float*) take((size_t)N * GAT_DHID * 4);

    const dim3 T(256);
    const int gNF   = (N * GAT_FIN  + 255) / 256;
    const int gNH   = (N * GAT_H    + 255) / 256;
    const int gND   = (N * GAT_DHID + 255) / 256;
    const int gEH   = (E * GAT_H    + 255) / 256;
    const int gGemm = (N + 127) / 128;

    // convert inputs / weights to bf16 (transposed weights for B fragments)
    k_cvt_bf16<<<gNF, T, 0, stream>>>(x, xb, N * GAT_FIN);
    k_cvt_w_t<<<(96 * GAT_FIN + 255) / 256, T, 0, stream>>>(W0, w0t, GAT_FIN);
    k_cvt_w_t<<<(96 * 96 + 255) / 256, T, 0, stream>>>(W1, w1t, 96);

    // ---- layer 0 (no activation) ----
    k_gemm_att<<<gGemm, T, 0, stream>>>(xb, w0t, hlin, N, GAT_FIN);
    k_alpha<<<gNH, T, 0, stream>>>(hlin, as0, ad0, asn, adn, N);
    k_init<<<gND, T, 0, stream>>>(agg, mbuf, zbuf, N);
    k_edge_max<<<gEH, T, 0, stream>>>(esrc, edst, asn, adn, mbuf, E);
    k_edge_sum<<<gEH, T, 0, stream>>>(esrc, edst, asn, adn, mbuf, zbuf, E);
    k_edge_agg<<<gEH, T, 0, stream>>>(esrc, edst, asn, adn, mbuf, zbuf, hlin, agg, E);
    k_bias_act<<<gND, T, 0, stream>>>(agg, b0, (float*)nullptr, h0b, 0, N);

    // ---- layer 1 (+ ReLU) ----
    k_gemm_att<<<gGemm, T, 0, stream>>>(h0b, w1t, hlin, N, 96);
    k_alpha<<<gNH, T, 0, stream>>>(hlin, as1, ad1, asn, adn, N);
    k_init<<<gND, T, 0, stream>>>(agg, mbuf, zbuf, N);
    k_edge_max<<<gEH, T, 0, stream>>>(esrc, edst, asn, adn, mbuf, E);
    k_edge_sum<<<gEH, T, 0, stream>>>(esrc, edst, asn, adn, mbuf, zbuf, E);
    k_edge_agg<<<gEH, T, 0, stream>>>(esrc, edst, asn, adn, mbuf, zbuf, hlin, agg, E);
    k_bias_act<<<gND, T, 0, stream>>>(agg, b1, h1, (__bf16*)nullptr, 1, N);

    // ---- classifier + log-softmax ----
    k_out<<<(N + OBLK - 1) / OBLK, dim3(OBLK), 0, stream>>>(h1, Wout, bout, out, N);
}
